// YvExpertChoiceMLP_6330781794495
// MI455X (gfx1250) — compile-verified, matrix operand order checked by reading, not run on hardware
//
#include <hip/hip_runtime.h>
#include <hip/hip_bf16.h>
#include <stdint.h>

#define B_ 4
#define S_ 2048
#define H_ 1024
#define I_ 4096
#define E_ 8
#define C_ 640   // int(min(S, (S*2//8)*1.25)) = 640

typedef __attribute__((ext_vector_type(16))) __bf16 bf16x16;
typedef __attribute__((ext_vector_type(8)))  float  f32x8;
typedef __attribute__((ext_vector_type(4)))  unsigned int u32x4;

#define WMMA_BF16(A, B, C) \
  __builtin_amdgcn_wmma_f32_16x16x32_bf16(false, (A), false, (B), (short)0, (C), false, false)

__device__ __forceinline__ unsigned short f2bf(float f) {
  unsigned int u = __builtin_bit_cast(unsigned int, f);
  u = (u + 0x7FFFu + ((u >> 16) & 1u)) >> 16;  // round-to-nearest-even
  return (unsigned short)u;
}

// Load one 16-bit WMMA operand fragment: per-lane two contiguous 16B K-chunks
// (K = base+half*8 .. +8 and K = base+16+half*8 .. +8), matching the ISA
// 16-bit A/B VGPR layout. Caller passes p already offset by row*stride + k0 + half*8.
__device__ __forceinline__ bf16x16 load_frag(const unsigned short* p) {
  union { u32x4 q[2]; bf16x16 v; } f;
  f.q[0] = *(const u32x4*)(p);
  f.q[1] = *(const u32x4*)(p + 16);
  return f.v;
}

// ---------------------------------------------------------------- zero init
__global__ void zero_kernel(float* __restrict__ p, long long n) {
  long long g = (long long)blockIdx.x * blockDim.x + threadIdx.x;
  if (g < n) p[g] = 0.0f;
}

// ------------------------------------------- f32 -> bf16 transposed weights
// src: [R][C] f32 per expert (blockIdx.z); dst: [C][R] bf16 per expert.
__global__ __launch_bounds__(256) void transpose_convert_kernel(
    const float* __restrict__ src, unsigned short* __restrict__ dst, int R, int C) {
  __shared__ float tile[32][33];
  const size_t es = (size_t)R * C;
  const float* s = src + (size_t)blockIdx.z * es;
  unsigned short* d = dst + (size_t)blockIdx.z * es;
  const int c0 = blockIdx.x * 32;
  const int r0 = blockIdx.y * 32;
  const int tx = threadIdx.x & 31;
  const int ty = threadIdx.x >> 5;  // 0..7
  #pragma unroll
  for (int yy = 0; yy < 32; yy += 8)
    tile[ty + yy][tx] = s[(size_t)(r0 + ty + yy) * C + c0 + tx];
  __syncthreads();
  #pragma unroll
  for (int yy = 0; yy < 32; yy += 8)
    d[(size_t)(c0 + ty + yy) * R + r0 + tx] = f2bf(tile[tx][ty + yy]);
}

// ------------------------------------------------------------------ router
// One 64-thread block per token: logits[t][0..7], plus softmax-usage and
// z-loss partial accumulation via f32 global atomics.
__global__ __launch_bounds__(64) void router_kernel(
    const float* __restrict__ x, const float* __restrict__ rw,
    float* __restrict__ logits, float* __restrict__ usage, float* __restrict__ zsum) {
  __shared__ float red[8][64];
  __shared__ float l8[8];
  const int t = blockIdx.x;
  const int tid = threadIdx.x;
  const float* xp = x + (size_t)t * H_;
  float acc[8];
  #pragma unroll
  for (int e = 0; e < 8; ++e) acc[e] = 0.0f;
  for (int h = tid; h < H_; h += 64) {
    float xv = xp[h];
    const float* rp = rw + h * E_;
    #pragma unroll
    for (int e = 0; e < 8; ++e) acc[e] += xv * rp[e];
  }
  #pragma unroll
  for (int e = 0; e < 8; ++e) red[e][tid] = acc[e];
  __syncthreads();
  if (tid < 8) {
    float s = 0.0f;
    #pragma unroll 8
    for (int i = 0; i < 64; ++i) s += red[tid][i];
    logits[(size_t)t * E_ + tid] = s;
    l8[tid] = s;
  }
  __syncthreads();
  if (tid == 0) {
    float mx = l8[0];
    #pragma unroll
    for (int e = 1; e < 8; ++e) mx = fmaxf(mx, l8[e]);
    float se = 0.0f, zz = 0.0f, ex[8];
    #pragma unroll
    for (int e = 0; e < 8; ++e) { ex[e] = __expf(l8[e] - mx); se += ex[e]; zz += l8[e] * l8[e]; }
    atomicAdd(zsum, zz);
    float inv = 1.0f / se;
    #pragma unroll
    for (int e = 0; e < 8; ++e) atomicAdd(&usage[e], ex[e] * inv);
  }
}

// ----------------------------------------------- per-(b,e) top-k + softmax
// Bitonic sort of the 2048 logits for one (b,e) in LDS (descending), then
// softmax over the top C_ values. Emits flat token indices (b*S + s).
__global__ __launch_bounds__(1024) void topk_kernel(
    const float* __restrict__ logits, int* __restrict__ topk_idx, float* __restrict__ topk_w) {
  __shared__ float v[S_];
  __shared__ int   ix[S_];
  __shared__ float red[1024];
  const int tid = threadIdx.x;
  const int be = blockIdx.x;
  const int b = be / E_;
  const int e = be % E_;
  for (int i = tid; i < S_; i += 1024) {
    v[i] = logits[(size_t)(b * S_ + i) * E_ + e];
    ix[i] = i;
  }
  __syncthreads();
  for (int k = 2; k <= S_; k <<= 1) {
    for (int j = k >> 1; j > 0; j >>= 1) {
      for (int i = tid; i < S_; i += 1024) {
        int l = i ^ j;
        if (l > i) {
          bool up = ((i & k) == 0);
          if ((v[i] < v[l]) == up) {  // descending overall
            float tv = v[i]; v[i] = v[l]; v[l] = tv;
            int ti = ix[i]; ix[i] = ix[l]; ix[l] = ti;
          }
        }
      }
      __syncthreads();
    }
  }
  const float mx = v[0];
  float p = (tid < C_) ? __expf(v[tid] - mx) : 0.0f;
  red[tid] = p;
  __syncthreads();
  for (int s = 512; s > 0; s >>= 1) {
    if (tid < s) red[tid] += red[tid + s];
    __syncthreads();
  }
  const float inv = 1.0f / red[0];
  if (tid < C_) {
    topk_w[(size_t)be * C_ + tid] = p * inv;
    topk_idx[(size_t)be * C_ + tid] = b * S_ + ix[tid];
  }
}

__global__ void counts_kernel(const int* __restrict__ topk_idx, float* __restrict__ counts) {
  int g = blockIdx.x * blockDim.x + threadIdx.x;
  if (g < B_ * E_ * C_) atomicAdd(&counts[topk_idx[g]], 1.0f);
}

// ------------------------------------------------------- fused expert MLP
// One workgroup (256 thr = 8 waves) per (b, e, 32-token tile).
// out_tile[32 x 1024] = silu(X[32 x 1024] @ w1[e]) @ w2[e], fused over I in
// 256-wide chunks so `mid` never leaves LDS. bf16 WMMA, f32 accumulate.
__global__ __launch_bounds__(256) void moe_mlp_kernel(
    const float* __restrict__ x,
    const unsigned short* __restrict__ w1t,   // [E][I][H] bf16 (transposed)
    const unsigned short* __restrict__ w2t,   // [E][H][I] bf16 (transposed)
    const int* __restrict__ topk_idx,         // [B*E][C] flat token id
    const float* __restrict__ topk_w,         // [B*E][C]
    float* __restrict__ out) {
  __shared__ unsigned short xtile[32 * H_];   // 64 KB bf16 activations
  __shared__ unsigned short midt[32 * 264];   // 16.5 KB silu(mid) chunk, padded stride
  __shared__ int   tokf[32];
  __shared__ float tokw[32];

  const int tid = threadIdx.x;
  const int mtb = blockIdx.x % (C_ / 32);
  const int be  = blockIdx.x / (C_ / 32);
  const int e   = be % E_;

  if (tid < 32) {
    int c = mtb * 32 + tid;
    tokf[tid] = topk_idx[(size_t)be * C_ + c];
    tokw[tid] = topk_w[(size_t)be * C_ + c];
  }
  __syncthreads();
  for (int idx = tid; idx < 32 * H_; idx += 256) {
    int row = idx >> 10;
    int col = idx & (H_ - 1);
    xtile[idx] = f2bf(x[(size_t)tokf[row] * H_ + col]);
  }
  __syncthreads();

  const int lane = tid & 31;
  const int wv = tid >> 5;       // wave 0..7
  const int r  = lane & 15;
  const int kh = lane >> 4;      // K-half per 16-bit operand layout

  const unsigned short* w1e = w1t + (size_t)e * ((size_t)I_ * H_);
  const unsigned short* w2e = w2t + (size_t)e * ((size_t)I_ * H_);

  const f32x8 zv = {0.f, 0.f, 0.f, 0.f, 0.f, 0.f, 0.f, 0.f};
  f32x8 c2[2][8];                 // persistent output accum: 32 x (wave's 128 cols)
  #pragma unroll
  for (int i = 0; i < 2; ++i)
    #pragma unroll
    for (int j = 0; j < 8; ++j) c2[i][j] = zv;

  for (int ic = 0; ic < I_; ic += 256) {
    // ---- phase 1: this wave's 32x32 slice of mid = X @ w1, K = H
    f32x8 c1[2][2];
    c1[0][0] = zv; c1[0][1] = zv; c1[1][0] = zv; c1[1][1] = zv;
    const int nb = ic + wv * 32;
    #pragma unroll 4
    for (int kk = 0; kk < H_; kk += 32) {
      bf16x16 a0 = load_frag(&xtile[(size_t)r * H_ + kk + kh * 8]);
      bf16x16 a1 = load_frag(&xtile[(size_t)(16 + r) * H_ + kk + kh * 8]);
      bf16x16 b0 = load_frag(w1e + (size_t)(nb + r) * H_ + kk + kh * 8);
      bf16x16 b1 = load_frag(w1e + (size_t)(nb + 16 + r) * H_ + kk + kh * 8);
      c1[0][0] = WMMA_BF16(a0, b0, c1[0][0]);
      c1[0][1] = WMMA_BF16(a0, b1, c1[0][1]);
      c1[1][0] = WMMA_BF16(a1, b0, c1[1][0]);
      c1[1][1] = WMMA_BF16(a1, b1, c1[1][1]);
    }
    __syncthreads();  // previous chunk's phase-2 readers done with midt
    #pragma unroll
    for (int mt = 0; mt < 2; ++mt)
      #pragma unroll
      for (int nt = 0; nt < 2; ++nt) {
        f32x8 vv = c1[mt][nt];
        #pragma unroll
        for (int vg = 0; vg < 8; ++vg) {
          float xv = vv[vg];
          float sv = xv / (1.0f + __expf(-xv));   // silu
          int m = mt * 16 + kh * 8 + vg;          // C/D layout: M = vgpr + half*8
          int n = wv * 32 + nt * 16 + r;          // N = lane&15
          midt[m * 264 + n] = f2bf(sv);
        }
      }
    __syncthreads();
    // ---- phase 2: out[:, wave's 128 cols] += silu(mid) @ w2, K = 256
    #pragma unroll 2
    for (int kk8 = 0; kk8 < 8; ++kk8) {
      int kb = kk8 * 32;
      bf16x16 a0 = load_frag(&midt[r * 264 + kb + kh * 8]);
      bf16x16 a1 = load_frag(&midt[(16 + r) * 264 + kb + kh * 8]);
      #pragma unroll
      for (int nt = 0; nt < 8; ++nt) {
        bf16x16 b = load_frag(w2e + (size_t)(wv * 128 + nt * 16 + r) * I_ + ic + kb + kh * 8);
        c2[0][nt] = WMMA_BF16(a0, b, c2[0][nt]);
        c2[1][nt] = WMMA_BF16(a1, b, c2[1][nt]);
      }
    }
    __syncthreads();
  }

  // ---- weighted scatter-add combine (tokens may be chosen by >1 expert)
  #pragma unroll
  for (int mt = 0; mt < 2; ++mt)
    #pragma unroll
    for (int nt = 0; nt < 8; ++nt) {
      f32x8 vv = c2[mt][nt];
      int n = wv * 128 + nt * 16 + r;
      #pragma unroll
      for (int vg = 0; vg < 8; ++vg) {
        int m = mt * 16 + kh * 8 + vg;
        atomicAdd(out + (size_t)tokf[m] * H_ + n, vv[vg] * tokw[m]);
      }
    }
}

__global__ void finalize_kernel(float* __restrict__ out, const float* __restrict__ counts) {
  size_t g = (size_t)blockIdx.x * blockDim.x + threadIdx.x;
  if (g < (size_t)B_ * S_ * H_) {
    size_t t = g >> 10;
    out[g] = out[g] / fmaxf(counts[t], 1.0f);
  }
}

__global__ void loss_kernel(const float* __restrict__ usage, const float* __restrict__ zsum,
                            float* __restrict__ outloss) {
  if (blockIdx.x == 0 && threadIdx.x == 0) {
    float aux = 0.0f;
    #pragma unroll
    for (int e = 0; e < 8; ++e) {
      float u = usage[e] * (1.0f / (float)(B_ * S_));
      float d = u - 1.0f / (float)E_;
      aux += d * d;                 // mean((u-1/E)^2)*E == sum(d^2)
    }
    float z = zsum[0] * (1.0f / (float)(B_ * S_ * E_));
    outloss[0] = 0.01f * aux + 0.001f * z;
  }
}

extern "C" void kernel_launch(void* const* d_in, const int* in_sizes, int n_in,
                              void* d_out, int out_size, void* d_ws, size_t ws_size,
                              hipStream_t stream) {
  const float* x  = (const float*)d_in[0];
  const float* rw = (const float*)d_in[1];
  const float* w1 = (const float*)d_in[2];
  const float* w2 = (const float*)d_in[3];
  float* out = (float*)d_out;

  char* ws = (char*)d_ws;
  size_t off = 0;
  unsigned short* w1t = (unsigned short*)(ws + off); off += (size_t)E_ * I_ * H_ * 2;   // 64 MB
  unsigned short* w2t = (unsigned short*)(ws + off); off += (size_t)E_ * I_ * H_ * 2;   // 64 MB
  float* logits = (float*)(ws + off); off += (size_t)B_ * S_ * E_ * 4;
  int*   tidx   = (int*)(ws + off);   off += (size_t)B_ * E_ * C_ * 4;
  float* tw     = (float*)(ws + off); off += (size_t)B_ * E_ * C_ * 4;
  float* counts = (float*)(ws + off); off += (size_t)B_ * S_ * 4;   // counts, usage, zsum contiguous
  float* usage  = (float*)(ws + off); off += 8 * 4;
  float* zsum   = (float*)(ws + off); off += 4;
  (void)ws_size; (void)in_sizes; (void)n_in; (void)out_size;

  {  // zero out (+loss slot)
    long long n = (long long)B_ * S_ * H_ + 1;
    zero_kernel<<<dim3((unsigned)((n + 255) / 256)), dim3(256), 0, stream>>>(out, n);
  }
  {  // zero counts + usage + zsum
    long long n = (long long)B_ * S_ + 9;
    zero_kernel<<<dim3((unsigned)((n + 255) / 256)), dim3(256), 0, stream>>>(counts, n);
  }
  // bf16 K-major weights (fit in 192 MB L2 -> tile re-reads stay on-chip)
  transpose_convert_kernel<<<dim3(I_ / 32, H_ / 32, E_), dim3(256), 0, stream>>>(w1, w1t, H_, I_);
  transpose_convert_kernel<<<dim3(H_ / 32, I_ / 32, E_), dim3(256), 0, stream>>>(w2, w2t, I_, H_);

  router_kernel<<<dim3(B_ * S_), dim3(64), 0, stream>>>(x, rw, logits, usage, zsum);
  topk_kernel<<<dim3(B_ * E_), dim3(1024), 0, stream>>>(logits, tidx, tw);
  counts_kernel<<<dim3((B_ * E_ * C_ + 255) / 256), dim3(256), 0, stream>>>(tidx, counts);

  moe_mlp_kernel<<<dim3(B_ * E_ * (C_ / 32)), dim3(256), 0, stream>>>(x, w1t, w2t, tidx, tw, out);

  finalize_kernel<<<dim3((B_ * S_ * H_ + 255) / 256), dim3(256), 0, stream>>>(out, counts);
  loss_kernel<<<dim3(1), dim3(1), 0, stream>>>(usage, zsum, out + (size_t)B_ * S_ * H_);
}
